// StandardAttention_74517682586460
// MI455X (gfx1250) — matrix operand order repaired and checked by reading, not compiled, not run on hardware
//
#include <hip/hip_runtime.h>
#include <hip/hip_bf16.h>
#include <math.h>

#define B_   2
#define S_   2048
#define D_   2048
#define H_   16
#define KVH_ 4
#define HD_  128
#define NEGV -1000000000.0f

typedef __bf16 v16bf __attribute__((ext_vector_type(16)));
typedef __bf16 v8bf  __attribute__((ext_vector_type(8)));
typedef float  v8f   __attribute__((ext_vector_type(8)));

// Pack two fp32 into two bf16 (round-with-bias) in ONE v_perm_b32 + 2 adds.
static __device__ __forceinline__ unsigned pk_bf16(float x, float y) {
    union { float f; unsigned u; } a, b;
    a.f = x; b.f = y;
    // dst = [x.hi16, y.hi16] ; sel bytes: 2,3 from src1(x), 6,7 from src0(y)
    return __builtin_amdgcn_perm(b.u + 0x8000u, a.u + 0x8000u, 0x07060302u);
}
// scalar RNE fp32 -> bf16 (cold paths)
static __device__ __forceinline__ __bf16 f2bf(float x) {
    union { float f; unsigned u; } v; v.f = x;
    unsigned b = v.u + 0x7FFFu + ((v.u >> 16) & 1u);
    unsigned short h = (unsigned short)(b >> 16);
    __bf16 r; __builtin_memcpy(&r, &h, 2); return r;
}

// Build a 16-element bf16 fragment from two contiguous 16-byte runs.
static __device__ __forceinline__ v16bf frag16(const __bf16* lo, const __bf16* hi) {
    v8bf a = *reinterpret_cast<const v8bf*>(lo);
    v8bf b = *reinterpret_cast<const v8bf*>(hi);
    v16bf r;
    __builtin_memcpy(&r, &a, 16);
    __builtin_memcpy(reinterpret_cast<char*>(&r) + 16, &b, 16);
    return r;
}
// A fragment (16x32, 16-bit): lane m holds k = half*8+0..7 and 16+half*8+0..7
static __device__ __forceinline__ v16bf frag_a(const __bf16* rowkk, int half) {
    return frag16(rowkk + half * 8, rowkk + 16 + half * 8);
}
// B fragment (32x16, 16-bit): lane n holds k = half*16+0..15 (contiguous 32B)
static __device__ __forceinline__ v16bf frag_b(const __bf16* rowkk, int half) {
    return frag16(rowkk + half * 16, rowkk + half * 16 + 8);
}

// CDNA5 async global->LDS copy (16 bytes/lane), ASYNCcnt-tracked.
static __device__ __forceinline__ void async_copy16(const __bf16* g, __bf16* l) {
    unsigned laddr = (unsigned)(uintptr_t)l;
    asm volatile("global_load_async_to_lds_b128 %0, %1, off"
                 :: "v"(laddr), "v"(g) : "memory");
}
static __device__ __forceinline__ void async_wait0() {
    asm volatile("s_wait_asynccnt 0x0" ::: "memory");
}

// ---------------------------------------------------------------------------
// fp32 -> bf16 bulk conversion (one-time staging). n must be a multiple of 8.
// ---------------------------------------------------------------------------
__global__ __launch_bounds__(256) void f32_to_bf16_vec(
    const float* __restrict__ src, __bf16* __restrict__ dst, size_t n)
{
    size_t i = ((size_t)blockIdx.x * 256 + threadIdx.x) * 8;
    if (i >= n) return;
    float4 x = *reinterpret_cast<const float4*>(src + i);
    float4 y = *reinterpret_cast<const float4*>(src + i + 4);
    unsigned o[4];
    o[0] = pk_bf16(x.x, x.y); o[1] = pk_bf16(x.z, x.w);
    o[2] = pk_bf16(y.x, y.y); o[3] = pk_bf16(y.z, y.w);
    *reinterpret_cast<uint4*>(dst + i) = *reinterpret_cast<uint4*>(o);
}

// ---------------------------------------------------------------------------
// GEMM: out[m,n] = sum_k A[m,k] * W[n,k]   (A,W bf16; out fp32)
// 256 thr = 8 waves; block tile 128(M) x 64(N); wave strip 16x64.
// B tile (64x32, 4KB) double-buffered in LDS via async global->LDS DMA.
// ---------------------------------------------------------------------------
__global__ __launch_bounds__(256) void gemm_bf16_wmma(
    const __bf16* __restrict__ A, const __bf16* __restrict__ W,
    float* __restrict__ out, int M, int N, int K)
{
    __shared__ __bf16 bt[2][64 * 32];
    const int tid  = threadIdx.x;
    const int lane = tid & 31;
    const int wave = tid >> 5;
    const int m    = lane & 15;
    const int half = lane >> 4;
    const int rowBase = blockIdx.y * 128 + wave * 16;
    const int colBase = blockIdx.x * 64;

    const __bf16* Arow = A + (size_t)(rowBase + m) * K;

    // cooperative staging: thread -> (col, 8-elem chunk of the 32-K slab)
    const int scol = tid >> 2;                 // 0..63
    const int skof = (tid & 3) * 8;            // 0,8,16,24
    const __bf16* gW = W + (size_t)(colBase + scol) * K + skof;
    __bf16* lb[2] = { &bt[0][scol * 32 + skof], &bt[1][scol * 32 + skof] };

    async_copy16(gW, lb[0]);
    async_wait0();
    __syncthreads();

    v8f c[4] = {};
    int buf = 0;
    for (int kk = 0; kk < K; kk += 32) {
        if (kk + 32 < K)
            async_copy16(gW + kk + 32, lb[buf ^ 1]);   // prefetch next slab
        if (kk + 64 < K) __builtin_prefetch(Arow + kk + 64, 0, 1);

        v16bf a = frag_a(Arow + kk, half);
        const __bf16* bbuf = bt[buf];
        #pragma unroll
        for (int t4 = 0; t4 < 4; ++t4) {
            const __bf16* brow = bbuf + (t4 * 16 + m) * 32 + half * 8;
            v16bf bfr = frag16(brow, brow + 16);
            c[t4] = __builtin_amdgcn_wmma_f32_16x16x32_bf16(
                false, a, false, bfr, (short)0, c[t4], false, false);
        }
        async_wait0();
        __syncthreads();
        buf ^= 1;
    }
    #pragma unroll
    for (int t4 = 0; t4 < 4; ++t4) {
        const int col = colBase + t4 * 16 + m;
        #pragma unroll
        for (int r = 0; r < 8; ++r)
            out[(size_t)(rowBase + r + half * 8) * N + col] = c[t4][r];
    }
}

// ---------------------------------------------------------------------------
// RoPE + pack fp32 [b,s,h,d] -> bf16 [b,h,s,d] (or transposed [b,h,d,s])
// ---------------------------------------------------------------------------
__global__ __launch_bounds__(256) void rope_pack(
    const float* __restrict__ src, __bf16* __restrict__ dst,
    const float* __restrict__ cosT, const float* __restrict__ sinT,
    int heads, int doRope, int transposed)
{
    const size_t idx = (size_t)blockIdx.x * blockDim.x + threadIdx.x;
    const size_t total = (size_t)B_ * S_ * heads * HD_;
    if (idx >= total) return;
    const int d  = (int)(idx % HD_);
    const int h  = (int)((idx / HD_) % heads);
    const int s  = (int)((idx / ((size_t)HD_ * heads)) % S_);
    const int bb = (int)(idx / ((size_t)HD_ * heads * S_));

    const size_t srcIdx = (((size_t)bb * S_ + s) * heads + h) * HD_ + d;
    float x = src[srcIdx];
    float val = x;
    if (doRope) {
        const size_t csIdx = ((size_t)bb * S_ + s) * HD_ + d;
        const float c  = cosT[csIdx];
        const float sn = sinT[csIdx];
        const float rot = (d < HD_ / 2) ? -src[srcIdx + HD_ / 2]
                                        :  src[srcIdx - HD_ / 2];
        val = x * c + rot * sn;
    }
    const size_t dstIdx = transposed
        ? (((size_t)bb * heads + h) * HD_ + d) * S_ + s
        : (((size_t)bb * heads + h) * S_ + s) * HD_ + d;
    dst[dstIdx] = f2bf(val);
}

// ---------------------------------------------------------------------------
// Scores: w[b,h,i,j] = scale * sum_d q*k + mask; causal-dead tiles get NEGV.
// K tile double-buffered in LDS (shared by all 8 row-waves of the block).
// ---------------------------------------------------------------------------
__global__ __launch_bounds__(256) void attn_score_wmma(
    const __bf16* __restrict__ qb, const __bf16* __restrict__ kb,
    const float* __restrict__ mask, float* __restrict__ weights, float scale)
{
    __shared__ __bf16 kt[2][64 * 32];
    const int tid  = threadIdx.x;
    const int lane = tid & 31;
    const int wave = tid >> 5;
    const int m    = lane & 15;
    const int half = lane >> 4;
    const int bh   = blockIdx.z;
    const int bb   = bh / H_;
    const int h    = bh % H_;
    const int kvh  = h / (H_ / KVH_);
    const int rowBase  = blockIdx.y * 128 + wave * 16;
    const int colBase0 = blockIdx.x * 64;

    float* wout = weights + (size_t)bh * S_ * S_;

    // whole block strictly above the diagonal -> pure NEGV fill, no barriers
    if (colBase0 > (int)blockIdx.y * 128 + 127) {
        #pragma unroll
        for (int t4 = 0; t4 < 4; ++t4) {
            const int col = colBase0 + t4 * 16 + m;
            #pragma unroll
            for (int r = 0; r < 8; ++r)
                wout[(size_t)(rowBase + r + half * 8) * S_ + col] = NEGV;
        }
        return;
    }

    int nlive = 0;                                // live 16-wide col tiles (per wave)
    if (rowBase + 15 >= colBase0) {
        nlive = ((rowBase + 15 - colBase0) >> 4) + 1;
        if (nlive > 4) nlive = 4;
    }

    const __bf16* qrow = qb + ((size_t)bh * S_ + rowBase + m) * HD_;
    const __bf16* kbh  = kb + (size_t)(bb * KVH_ + kvh) * S_ * HD_;

    const int scol = tid >> 2;
    const int skof = (tid & 3) * 8;
    const __bf16* gK = kbh + (size_t)(colBase0 + scol) * HD_ + skof;
    __bf16* lb[2] = { &kt[0][scol * 32 + skof], &kt[1][scol * 32 + skof] };

    async_copy16(gK, lb[0]);
    async_wait0();
    __syncthreads();

    v8f c[4] = {};
    int buf = 0;
    #pragma unroll
    for (int kk = 0; kk < HD_; kk += 32) {
        if (kk + 32 < HD_)
            async_copy16(gK + kk + 32, lb[buf ^ 1]);
        if (nlive) {                              // wave-uniform
            v16bf a = frag_a(qrow + kk, half);
            const __bf16* kbuf = kt[buf];
            #pragma unroll
            for (int t4 = 0; t4 < 4; ++t4) {
                if (t4 >= nlive) continue;
                const __bf16* krow = kbuf + (t4 * 16 + m) * 32 + half * 8;
                v16bf bfr = frag16(krow, krow + 16);
                c[t4] = __builtin_amdgcn_wmma_f32_16x16x32_bf16(
                    false, a, false, bfr, (short)0, c[t4], false, false);
            }
        }
        async_wait0();
        __syncthreads();
        buf ^= 1;
    }
    #pragma unroll
    for (int t4 = 0; t4 < 4; ++t4) {
        const int col = colBase0 + t4 * 16 + m;
        #pragma unroll
        for (int r = 0; r < 8; ++r) {
            const int row = rowBase + r + half * 8;
            float v = NEGV;
            if (t4 < nlive)
                v = c[t4][r] * scale + mask[((size_t)bb * S_ + row) * S_ + col];
            wout[(size_t)row * S_ + col] = v;
        }
    }
}

// ---------------------------------------------------------------------------
// Row softmax in place. Row (8 KB) staged into LDS via async global->LDS DMA.
// ---------------------------------------------------------------------------
__global__ __launch_bounds__(256) void softmax_rows(float* __restrict__ w)
{
    __shared__ float rowbuf[S_];
    __shared__ float red[256];
    float* p = w + (size_t)blockIdx.x * S_;
    const int tid = threadIdx.x;

    {
        unsigned lds0   = (unsigned)(uintptr_t)&rowbuf[tid * 8];
        const float* g0 = p + tid * 8;
        asm volatile("global_load_async_to_lds_b128 %0, %1, off"
                     :: "v"(lds0), "v"(g0) : "memory");
        asm volatile("global_load_async_to_lds_b128 %0, %1, off offset:16"
                     :: "v"(lds0), "v"(g0) : "memory");
        asm volatile("s_wait_asynccnt 0x0" ::: "memory");
    }
    __syncthreads();

    float vals[8];
    float mx = -INFINITY;
    #pragma unroll
    for (int j = 0; j < 8; ++j) {
        vals[j] = rowbuf[tid + j * 256];
        mx = fmaxf(mx, vals[j]);
    }
    red[tid] = mx; __syncthreads();
    for (int s = 128; s > 0; s >>= 1) {
        if (tid < s) red[tid] = fmaxf(red[tid], red[tid + s]);
        __syncthreads();
    }
    mx = red[0]; __syncthreads();

    float sum = 0.f;
    #pragma unroll
    for (int j = 0; j < 8; ++j) { vals[j] = __expf(vals[j] - mx); sum += vals[j]; }
    red[tid] = sum; __syncthreads();
    for (int s = 128; s > 0; s >>= 1) {
        if (tid < s) red[tid] += red[tid + s];
        __syncthreads();
    }
    const float inv = 1.0f / red[0];
    #pragma unroll
    for (int j = 0; j < 8; ++j) p[tid + j * 256] = vals[j] * inv;
}

// ---------------------------------------------------------------------------
// AV: ctx[b,i,h*HD+n] = sum_k w[b,h,i,k] * v[b,kvh,k,n]
// V (K-major [b,kvh,d,s]) tile double-buffered in LDS; causal K truncation.
// ---------------------------------------------------------------------------
__global__ __launch_bounds__(256) void attn_av_wmma(
    const float* __restrict__ w, const __bf16* __restrict__ vbT,
    float* __restrict__ ctx)
{
    __shared__ __bf16 vt2[2][64 * 32];
    const int tid  = threadIdx.x;
    const int lane = tid & 31;
    const int wave = tid >> 5;
    const int m    = lane & 15;
    const int half = lane >> 4;
    const int bh   = blockIdx.z;
    const int bb   = bh / H_;
    const int h    = bh % H_;
    const int kvh  = h / (H_ / KVH_);
    const int rowBase = blockIdx.y * 128 + wave * 16;
    const int colBase = blockIdx.x * 64;

    const float*  wrow = w + ((size_t)bh * S_ + rowBase + m) * S_;
    const __bf16* vt   = vbT + (size_t)(bb * KVH_ + kvh) * HD_ * S_;

    const int scol = tid >> 2;
    const int skof = (tid & 3) * 8;
    const __bf16* gV = vt + (size_t)(colBase + scol) * S_ + skof;
    __bf16* lb[2] = { &vt2[0][scol * 32 + skof], &vt2[1][scol * 32 + skof] };

    const int blockKmax = (int)blockIdx.y * 128 + 128;   // longest wave in block
    const int kmaxWave  = rowBase + 16;                  // this wave's causal bound

    async_copy16(gV, lb[0]);
    async_wait0();
    __syncthreads();

    v8f c[4] = {};
    int buf = 0;
    for (int kk = 0; kk < blockKmax; kk += 32) {
        if (kk + 32 < blockKmax)
            async_copy16(gV + kk + 32, lb[buf ^ 1]);
        if (kk < kmaxWave) {                             // wave-uniform
            if (kk + 64 < kmaxWave) __builtin_prefetch(wrow + kk + 64, 0, 1);
            v16bf a;
            #pragma unroll
            for (int r = 0; r < 8; ++r) {
                const int kOff = ((r >> 2) << 4) + (half << 3) + ((r & 3) << 1);
                float2 t = *reinterpret_cast<const float2*>(wrow + kk + kOff);
                unsigned p = pk_bf16(t.x, t.y);
                __builtin_memcpy(reinterpret_cast<char*>(&a) + 4 * r, &p, 4);
            }
            const __bf16* vbuf = vt2[buf];
            #pragma unroll
            for (int t4 = 0; t4 < 4; ++t4) {
                const __bf16* vrow = vbuf + (t4 * 16 + m) * 32 + half * 8;
                v16bf bfr = frag16(vrow, vrow + 16);
                c[t4] = __builtin_amdgcn_wmma_f32_16x16x32_bf16(
                    false, a, false, bfr, (short)0, c[t4], false, false);
            }
        }
        async_wait0();
        __syncthreads();
        buf ^= 1;
    }
    #pragma unroll
    for (int t4 = 0; t4 < 4; ++t4) {
        const int col = colBase + t4 * 16 + m;
        #pragma unroll
        for (int r = 0; r < 8; ++r) {
            const int row = rowBase + r + half * 8;
            ctx[((size_t)bb * S_ + row) * D_ + h * HD_ + col] = c[t4][r];
        }
    }
}

// ---------------------------------------------------------------------------
extern "C" void kernel_launch(void* const* d_in, const int* in_sizes, int n_in,
                              void* d_out, int out_size, void* d_ws, size_t ws_size,
                              hipStream_t stream)
{
    const float* hidden = (const float*)d_in[0];
    const float* cosT   = (const float*)d_in[1];
    const float* sinT   = (const float*)d_in[2];
    const float* mask   = (const float*)d_in[3];
    const float* Wq     = (const float*)d_in[4];
    const float* Wk     = (const float*)d_in[5];
    const float* Wv     = (const float*)d_in[6];
    const float* Wo     = (const float*)d_in[7];

    float* attn_out = (float*)d_out;                       // [B,S,D]
    float* weights  = attn_out + (size_t)B_ * S_ * D_;     // [B,H,S,S]

    const size_t nH  = (size_t)B_ * S_ * H_ * HD_;         // 8.4M (== B*S*D)
    const size_t nKV = (size_t)B_ * S_ * KVH_ * HD_;       // 2.1M
    const size_t nWq = (size_t)H_ * HD_ * D_;              // 4.2M
    const size_t nWk = (size_t)KVH_ * HD_ * D_;            // 1.05M

    char* wsb = (char*)d_ws;
    size_t off = 0;
    float*  qp   = (float*)(wsb + off); off += nH  * sizeof(float);
    float*  kp   = (float*)(wsb + off); off += nKV * sizeof(float);
    float*  vp   = (float*)(wsb + off); off += nKV * sizeof(float);
    float*  ctx  = (float*)(wsb + off); off += nH  * sizeof(float);
    __bf16* hb   = (__bf16*)(wsb + off); off += nH  * sizeof(__bf16);
    __bf16* wqb  = (__bf16*)(wsb + off); off += nWq * sizeof(__bf16);
    __bf16* wkb  = (__bf16*)(wsb + off); off += nWk * sizeof(__bf16);
    __bf16* wvb  = (__bf16*)(wsb + off); off += nWk * sizeof(__bf16);
    __bf16* wob  = (__bf16*)(wsb + off); off += nWq * sizeof(__bf16);
    __bf16* qb   = (__bf16*)(wsb + off); off += nH  * sizeof(__bf16);
    __bf16* kbf  = (__bf16*)(wsb + off); off += nKV * sizeof(__bf16);
    __bf16* vbT  = (__bf16*)(wsb + off); off += nKV * sizeof(__bf16);
    __bf16* ctxb = (__bf16*)(wsb + off); off += nH  * sizeof(__bf16);

    const dim3 blk(256);
    const int M = B_ * S_;                                 // 4096

    // one-time bf16 staging of GEMM operands
    f32_to_bf16_vec<<<dim3((unsigned)(nH  / 2048)), blk, 0, stream>>>(hidden, hb,  nH);
    f32_to_bf16_vec<<<dim3((unsigned)(nWq / 2048)), blk, 0, stream>>>(Wq,     wqb, nWq);
    f32_to_bf16_vec<<<dim3((unsigned)(nWk / 2048)), blk, 0, stream>>>(Wk,     wkb, nWk);
    f32_to_bf16_vec<<<dim3((unsigned)(nWk / 2048)), blk, 0, stream>>>(Wv,     wvb, nWk);
    f32_to_bf16_vec<<<dim3((unsigned)(nWq / 2048)), blk, 0, stream>>>(Wo,     wob, nWq);

    // QKV projections (bf16 WMMA, fp32 accumulate)
    gemm_bf16_wmma<<<dim3(D_ / 64, M / 128), blk, 0, stream>>>(hb, wqb, qp, M, H_ * HD_,   D_);
    gemm_bf16_wmma<<<dim3((KVH_ * HD_) / 64, M / 128), blk, 0, stream>>>(hb, wkb, kp, M, KVH_ * HD_, D_);
    gemm_bf16_wmma<<<dim3((KVH_ * HD_) / 64, M / 128), blk, 0, stream>>>(hb, wvb, vp, M, KVH_ * HD_, D_);

    // RoPE + bf16 pack (V transposed to K-major for contiguous B fragments)
    rope_pack<<<dim3((unsigned)((nH  + 255) / 256)), blk, 0, stream>>>(qp, qb,  cosT, sinT, H_,   1, 0);
    rope_pack<<<dim3((unsigned)((nKV + 255) / 256)), blk, 0, stream>>>(kp, kbf, cosT, sinT, KVH_, 1, 0);
    rope_pack<<<dim3((unsigned)((nKV + 255) / 256)), blk, 0, stream>>>(vp, vbT, cosT, sinT, KVH_, 0, 1);

    // scores + mask -> weights (part of d_out)
    const float scale = 0.08838834764831845f;              // 128^-0.5
    attn_score_wmma<<<dim3(S_ / 64, S_ / 128, B_ * H_), blk, 0, stream>>>(qb, kbf, mask, weights, scale);

    // softmax in place (async global->LDS staging)
    softmax_rows<<<dim3(B_ * H_ * S_), blk, 0, stream>>>(weights);

    // attn @ V -> ctx [B,S,H*HD]
    attn_av_wmma<<<dim3(HD_ / 64, S_ / 128, B_ * H_), blk, 0, stream>>>(weights, vbT, ctx);

    // stage ctx to bf16, then output projection
    f32_to_bf16_vec<<<dim3((unsigned)(nH / 2048)), blk, 0, stream>>>(ctx, ctxb, nH);
    gemm_bf16_wmma<<<dim3(D_ / 64, M / 128), blk, 0, stream>>>(ctxb, wob, attn_out, M, D_, D_);
}
